// MultiheadSelfAttention_16466904613796
// MI455X (gfx1250) — compile-verified, matrix operand order checked by reading, not run on hardware
//
#include <hip/hip_runtime.h>

typedef __bf16 bf16;
typedef __attribute__((ext_vector_type(16))) __bf16 v16bf;
typedef __attribute__((ext_vector_type(8)))  __bf16 v8bf;
typedef __attribute__((ext_vector_type(8)))  float   v8f;
typedef __attribute__((ext_vector_type(4)))  unsigned int v4u;
typedef __attribute__((ext_vector_type(8)))  int v8i_t;
typedef __attribute__((ext_vector_type(4)))  int v4i_t;

#define D_MODEL 1024
#define NUM_HEADS 16
#define D_K 64
#define SEQ 2048
#define BATCH 2
#define M_ROWS (BATCH * SEQ)   // 4096
#define LN10000_OVER_64 0.14391157f

// ---------------------------------------------------------------------------
// WMMA wrapper: D = A(16x32 bf16) * B(32x16 bf16) + C(16x16 f32)
// ---------------------------------------------------------------------------
__device__ __forceinline__ v8f wmma_bf16(v16bf a, v16bf b, v8f c) {
  return __builtin_amdgcn_wmma_f32_16x16x32_bf16(
      /*neg_a=*/false, a, /*neg_b=*/false, b,
      /*c_mod=*/(short)0, c, /*reuse_a=*/false, /*reuse_b=*/false);
}

// ---------------------------------------------------------------------------
// 16x32 bf16 fragment from a row-major source (global or LDS; addrspace is
// inferred after inlining).  ISA layout (CDNA5 ISA, 16-bit A 16x32):
//   lanes 0-15 : row M=lane,    K = {0..7} and {16..23}
//   lanes 16-31: row M=lane-16, K = {8..15} and {24..31}
// ---------------------------------------------------------------------------
__device__ __forceinline__ v16bf load_frag(const bf16* base, int ld,
                                           int row0, int k0) {
  int lane = threadIdx.x & 31;
  int l15 = lane & 15, hi = lane >> 4;
  const bf16* p = base + (size_t)(row0 + l15) * ld + k0 + hi * 8;
  v8bf lo = *(const v8bf*)p;
  v8bf hh = *(const v8bf*)(p + 16);
  v16bf f;
#pragma unroll
  for (int i = 0; i < 8; ++i) { f[i] = lo[i]; f[i + 8] = hh[i]; }
  return f;
}

// ---------------------------------------------------------------------------
// Tensor Data Mover: 2D bf16 tile (tile_d0 x tile_d1 rows) global -> LDS.
// D# packing per CDNA5 ISA (group0: count/lds/global/
// type=2; group1: data_size=2B, tensor dims, tile dims, dim0 stride).
// ---------------------------------------------------------------------------
__device__ __forceinline__ void tdm_load_2d_bf16(
    unsigned lds_off, const bf16* gptr,
    unsigned tile_d0, unsigned tile_d1,
    unsigned tensor_d0, unsigned tensor_d1, unsigned stride_elems) {
  unsigned long long ga = (unsigned long long)(uintptr_t)gptr;
  v4u g0;
  g0[0] = 1u;                                        // count=1, user D#
  g0[1] = lds_off;                                   // lds_addr [63:32]
  g0[2] = (unsigned)(ga & 0xffffffffu);              // global_addr lo
  g0[3] = (unsigned)((ga >> 32) & 0x01ffffffu)       // global_addr hi
        | 0x80000000u;                               // type=2 ("image")
  v8i_t g1;
  g1[0] = (int)(1u << 16);                           // data_size=1 (2 bytes)
  g1[1] = (int)(tensor_d0 << 16);                    // tensor_dim0[15:0]
  g1[2] = (int)((tensor_d0 >> 16) | (tensor_d1 << 16));
  g1[3] = (int)((tensor_d1 >> 16) | (tile_d0 << 16)); // tile_dim0
  g1[4] = (int)(tile_d1 & 0xffffu);                  // tile_dim1 (tile_dim2=0)
  g1[5] = (int)stride_elems;                         // tensor_dim0_stride lo32
  g1[6] = 0;
  g1[7] = 0;
  v4i_t z4 = {0, 0, 0, 0};
#if __clang_major__ >= 23
  v8i_t z8 = {0, 0, 0, 0, 0, 0, 0, 0};
  __builtin_amdgcn_tensor_load_to_lds(g0, g1, z4, z4, z8, 0);
#else
  __builtin_amdgcn_tensor_load_to_lds(g0, g1, z4, z4, 0);
#endif
}

// ---------------------------------------------------------------------------
// fp32 -> bf16 conversion (all sizes are multiples of 4)
// ---------------------------------------------------------------------------
__global__ void cvt_f32_to_bf16(const float* __restrict__ in,
                                bf16* __restrict__ out, int n) {
  int i = (blockIdx.x * blockDim.x + threadIdx.x) * 4;
  if (i < n) {
    float4 v = *(const float4*)(in + i);
    out[i + 0] = (bf16)v.x;
    out[i + 1] = (bf16)v.y;
    out[i + 2] = (bf16)v.z;
    out[i + 3] = (bf16)v.w;
  }
}

// ---------------------------------------------------------------------------
// QKV projection: out[row,col] = sum_d X[row,d] * W[col,d]   (X @ W^T)
// Each wave owns a 32x64 output tile: 2 A-frags x 4 B-frags -> 8 WMMA per
// 32-wide k step (fragment reuse ~21 FLOP/B from L2).
// Epilogue: optional fused RoPE (Q,K), optional transposed store (V).
// ---------------------------------------------------------------------------
__global__ __launch_bounds__(256) void qkv_gemm(
    const bf16* __restrict__ X, const bf16* __restrict__ W,
    const int* __restrict__ tpos, bf16* __restrict__ out,
    int do_rope, int transposeV) {
  int wave = threadIdx.x >> 5;
  int lane = threadIdx.x & 31;
  int l15 = lane & 15, hi = lane >> 4;
  int tile = blockIdx.x * 8 + wave;
  const int NT = D_MODEL / 64;          // 16 column tiles
  int tm = (tile / NT) * 32;
  int tn = (tile % NT) * 64;

  v8f acc[2][4] = {};
  for (int k = 0; k < D_MODEL; k += 32) {
    v16bf a0 = load_frag(X, D_MODEL, tm, k);
    v16bf a1 = load_frag(X, D_MODEL, tm + 16, k);
    v16bf b0 = load_frag(W, D_MODEL, tn, k);
    v16bf b1 = load_frag(W, D_MODEL, tn + 16, k);
    v16bf b2 = load_frag(W, D_MODEL, tn + 32, k);
    v16bf b3 = load_frag(W, D_MODEL, tn + 48, k);
    acc[0][0] = wmma_bf16(a0, b0, acc[0][0]);
    acc[0][1] = wmma_bf16(a0, b1, acc[0][1]);
    acc[0][2] = wmma_bf16(a0, b2, acc[0][2]);
    acc[0][3] = wmma_bf16(a0, b3, acc[0][3]);
    acc[1][0] = wmma_bf16(a1, b0, acc[1][0]);
    acc[1][1] = wmma_bf16(a1, b1, acc[1][1]);
    acc[1][2] = wmma_bf16(a1, b2, acc[1][2]);
    acc[1][3] = wmma_bf16(a1, b3, acc[1][3]);
  }

#pragma unroll
  for (int r = 0; r < 2; ++r) {
#pragma unroll
    for (int c = 0; c < 4; ++c) {
      int col = tn + c * 16 + l15;      // output feature 0..1023
      int h = col >> 6, d = col & 63;
      float fr = 0.0f;
      if (do_rope) fr = __expf(-LN10000_OVER_64 * (float)(d & ~1));
#pragma unroll
      for (int i = 0; i < 8; ++i) {
        int row = tm + r * 16 + i + 8 * hi;   // flat b*SEQ + t
        int bb = row >> 11, t = row & (SEQ - 1);
        float v = acc[r][c][i];
        if (do_rope) {
          float partner = __shfl_xor(v, 1);
          float ang = (float)tpos[row] * fr;
          float sn, cs;
          __sincosf(ang, &sn, &cs);
          v = (col & 1) ? (partner * sn + v * cs) : (v * cs - partner * sn);
        }
        if (transposeV) {
          out[(((size_t)bb * NUM_HEADS + h) * D_K + d) * SEQ + t] = (bf16)v;
        } else {
          out[(((size_t)bb * NUM_HEADS + h) * SEQ + t) * D_K + d] = (bf16)v;
        }
      }
    }
  }
}

// ---------------------------------------------------------------------------
// Flash attention: grid (B*H, S/128), 8 waves/block, each wave owns 16 q rows.
// Per 32-wide k tile: wave 0 TDM-loads the 32x64 K tile and 64x32 V^T tile
// into block-shared LDS (TENSORcnt + barriers), then every wave runs
// 4 WMMAs for Q*K^T, online softmax, P re-layout via per-wave LDS, and
// 4 WMMAs for P*V.
// ---------------------------------------------------------------------------
__global__ __launch_bounds__(256) void flash_attn(
    const bf16* __restrict__ Qb, const bf16* __restrict__ Kb,
    const bf16* __restrict__ Vtb, bf16* __restrict__ attnb) {
  __shared__ __attribute__((aligned(16))) bf16 ldsK[32 * 64];  // [k][d] 4KB
  __shared__ __attribute__((aligned(16))) bf16 ldsV[64 * 32];  // [d][k] 4KB
  __shared__ __attribute__((aligned(16))) bf16 pbuf[8][16 * 32];

  int wave = threadIdx.x >> 5;
  int lane = threadIdx.x & 31;
  int l15 = lane & 15, hi = lane >> 4;
  int bh = blockIdx.x;                  // b*NUM_HEADS + h
  int q0 = blockIdx.y * 128 + wave * 16;
  int b = bh >> 4, h = bh & 15;

  const bf16* Q  = Qb  + (size_t)bh * SEQ * D_K;
  const bf16* K  = Kb  + (size_t)bh * SEQ * D_K;
  const bf16* Vt = Vtb + (size_t)bh * D_K * SEQ;

  v16bf aQ0 = load_frag(Q, D_K, q0, 0);
  v16bf aQ1 = load_frag(Q, D_K, q0, 32);

  v8f o0 = {}, o1 = {}, o2 = {}, o3 = {};
  float m[8], l[8];
#pragma unroll
  for (int i = 0; i < 8; ++i) { m[i] = -1e30f; l[i] = 0.0f; }

  unsigned ldsK_off = (unsigned)(uintptr_t)(void*)&ldsK[0];
  unsigned ldsV_off = (unsigned)(uintptr_t)(void*)&ldsV[0];

  int myntiles = (q0 + 15) / 32 + 1;         // causal: k <= q0+15
  int ntiles_max = 4 * (blockIdx.y + 1);     // tiles for the block's last wave

  for (int kt = 0; kt < ntiles_max; ++kt) {
    int k0 = kt * 32;

    // All waves done reading the previous K/V tiles before TDM overwrites.
    asm volatile("s_wait_dscnt 0x0" ::: "memory");
    __syncthreads();
    if (wave == 0) {
      // K tile: 32 rows x 64 cols, row stride 64 (contiguous 4KB)
      tdm_load_2d_bf16(ldsK_off, K + (size_t)k0 * D_K,
                       /*tile_d0=*/64, /*tile_d1=*/32,
                       /*tensor_d0=*/64, /*tensor_d1=*/SEQ, /*stride=*/64);
      // V^T tile: 64 rows (d) x 32 cols (k), row stride SEQ
      tdm_load_2d_bf16(ldsV_off, Vt + k0,
                       /*tile_d0=*/32, /*tile_d1=*/64,
                       /*tensor_d0=*/SEQ, /*tensor_d1=*/D_K, /*stride=*/SEQ);
      __builtin_amdgcn_s_wait_tensorcnt(0);
    }
    __syncthreads();

    if (kt < myntiles) {
      v8f z = {};
      v8f s0 = wmma_bf16(aQ0, load_frag(ldsK, D_K, 0, 0), z);
      s0     = wmma_bf16(aQ1, load_frag(ldsK, D_K, 0, 32), s0);
      v8f s1 = wmma_bf16(aQ0, load_frag(ldsK, D_K, 16, 0), z);
      s1     = wmma_bf16(aQ1, load_frag(ldsK, D_K, 16, 32), s1);

#pragma unroll
      for (int i = 0; i < 8; ++i) {
        int qrow = q0 + i + 8 * hi;
        float x0 = s0[i] * 0.125f;      // 1/sqrt(64)
        float x1 = s1[i] * 0.125f;
        if (k0 + l15 > qrow)      x0 = -1e30f;
        if (k0 + 16 + l15 > qrow) x1 = -1e30f;

        float mx = fmaxf(x0, x1);
#pragma unroll
        for (int off = 8; off > 0; off >>= 1)
          mx = fmaxf(mx, __shfl_xor(mx, off, 16));

        float mn = fmaxf(m[i], mx);
        float a  = __expf(m[i] - mn);
        m[i] = mn;
        float p0 = __expf(x0 - mn);
        float p1 = __expf(x1 - mn);
        l[i] = l[i] * a + p0 + p1;      // per-lane partial row sum
        o0[i] *= a; o1[i] *= a; o2[i] *= a; o3[i] *= a;

        pbuf[wave][(i + 8 * hi) * 32 + l15]      = (bf16)p0;
        pbuf[wave][(i + 8 * hi) * 32 + l15 + 16] = (bf16)p1;
      }

      asm volatile("s_wait_dscnt 0x0" ::: "memory");  // P visible to reloads

      v16bf aP = load_frag(&pbuf[wave][0], 32, 0, 0);
      o0 = wmma_bf16(aP, load_frag(ldsV, 32, 0,  0), o0);
      o1 = wmma_bf16(aP, load_frag(ldsV, 32, 16, 0), o1);
      o2 = wmma_bf16(aP, load_frag(ldsV, 32, 32, 0), o2);
      o3 = wmma_bf16(aP, load_frag(ldsV, 32, 48, 0), o3);
    }
  }

#pragma unroll
  for (int i = 0; i < 8; ++i) {
    float li = l[i];
#pragma unroll
    for (int off = 8; off > 0; off >>= 1)
      li += __shfl_xor(li, off, 16);
    float inv = 1.0f / li;
    int t = q0 + i + 8 * hi;
    size_t rb = ((size_t)b * SEQ + t) * D_MODEL + h * D_K;
    attnb[rb + l15]      = (bf16)(o0[i] * inv);
    attnb[rb + 16 + l15] = (bf16)(o1[i] * inv);
    attnb[rb + 32 + l15] = (bf16)(o2[i] * inv);
    attnb[rb + 48 + l15] = (bf16)(o3[i] * inv);
  }
}

// ---------------------------------------------------------------------------
// Output projection: out = attn @ Wo^T, fp32 store. 32x64 tile per wave.
// ---------------------------------------------------------------------------
__global__ __launch_bounds__(256) void out_proj(
    const bf16* __restrict__ A, const bf16* __restrict__ Wo,
    float* __restrict__ out) {
  int wave = threadIdx.x >> 5;
  int lane = threadIdx.x & 31;
  int l15 = lane & 15, hi = lane >> 4;
  int tile = blockIdx.x * 8 + wave;
  const int NT = D_MODEL / 64;
  int tm = (tile / NT) * 32, tn = (tile % NT) * 64;

  v8f acc[2][4] = {};
  for (int k = 0; k < D_MODEL; k += 32) {
    v16bf a0 = load_frag(A, D_MODEL, tm, k);
    v16bf a1 = load_frag(A, D_MODEL, tm + 16, k);
    v16bf b0 = load_frag(Wo, D_MODEL, tn, k);
    v16bf b1 = load_frag(Wo, D_MODEL, tn + 16, k);
    v16bf b2 = load_frag(Wo, D_MODEL, tn + 32, k);
    v16bf b3 = load_frag(Wo, D_MODEL, tn + 48, k);
    acc[0][0] = wmma_bf16(a0, b0, acc[0][0]);
    acc[0][1] = wmma_bf16(a0, b1, acc[0][1]);
    acc[0][2] = wmma_bf16(a0, b2, acc[0][2]);
    acc[0][3] = wmma_bf16(a0, b3, acc[0][3]);
    acc[1][0] = wmma_bf16(a1, b0, acc[1][0]);
    acc[1][1] = wmma_bf16(a1, b1, acc[1][1]);
    acc[1][2] = wmma_bf16(a1, b2, acc[1][2]);
    acc[1][3] = wmma_bf16(a1, b3, acc[1][3]);
  }
#pragma unroll
  for (int r = 0; r < 2; ++r) {
#pragma unroll
    for (int c = 0; c < 4; ++c) {
#pragma unroll
      for (int i = 0; i < 8; ++i) {
        int row = tm + r * 16 + i + 8 * hi;
        out[(size_t)row * D_MODEL + tn + c * 16 + l15] = acc[r][c][i];
      }
    }
  }
}

// ---------------------------------------------------------------------------
extern "C" void kernel_launch(void* const* d_in, const int* in_sizes, int n_in,
                              void* d_out, int out_size, void* d_ws,
                              size_t ws_size, hipStream_t stream) {
  const float* x  = (const float*)d_in[0];
  const int*   tp = (const int*)d_in[1];
  const float* Wq = (const float*)d_in[2];
  const float* Wk = (const float*)d_in[3];
  const float* Wv = (const float*)d_in[4];
  const float* Wo = (const float*)d_in[5];

  char* ws = (char*)d_ws;
  const size_t MB = 1024 * 1024;
  bf16* xb  = (bf16*)(ws + 0);        //  8 MB  x in bf16
  bf16* wqb = (bf16*)(ws + 8 * MB);   //  2 MB
  bf16* wkb = (bf16*)(ws + 10 * MB);  //  2 MB
  bf16* wvb = (bf16*)(ws + 12 * MB);  //  2 MB
  bf16* wob = (bf16*)(ws + 14 * MB);  //  2 MB
  bf16* qb  = (bf16*)(ws + 16 * MB);  //  8 MB  Q  [b][h][s][dk] (post-RoPE)
  bf16* kb  = (bf16*)(ws + 24 * MB);  //  8 MB  K  [b][h][s][dk] (post-RoPE)
  bf16* vtb = (bf16*)(ws + 32 * MB);  //  8 MB  V^T[b][h][dk][s]
  bf16* ab  = (bf16*)(ws + 40 * MB);  //  8 MB  attn out [b][s][1024]

  const int nX = M_ROWS * D_MODEL;      // 4194304
  const int nW = D_MODEL * D_MODEL;     // 1048576

  cvt_f32_to_bf16<<<(nX / 4 + 255) / 256, 256, 0, stream>>>(x, xb, nX);
  cvt_f32_to_bf16<<<(nW / 4 + 255) / 256, 256, 0, stream>>>(Wq, wqb, nW);
  cvt_f32_to_bf16<<<(nW / 4 + 255) / 256, 256, 0, stream>>>(Wk, wkb, nW);
  cvt_f32_to_bf16<<<(nW / 4 + 255) / 256, 256, 0, stream>>>(Wv, wvb, nW);
  cvt_f32_to_bf16<<<(nW / 4 + 255) / 256, 256, 0, stream>>>(Wo, wob, nW);

  const int tiles  = (M_ROWS / 32) * (D_MODEL / 64);  // 2048 wave tiles
  const int blocks = tiles / 8;                       // 256 (8 waves/block)

  qkv_gemm<<<blocks, 256, 0, stream>>>(xb, wqb, tp, qb,  1, 0);  // Q + RoPE
  qkv_gemm<<<blocks, 256, 0, stream>>>(xb, wkb, tp, kb,  1, 0);  // K + RoPE
  qkv_gemm<<<blocks, 256, 0, stream>>>(xb, wvb, tp, vtb, 0, 1);  // V^T

  flash_attn<<<dim3(BATCH * NUM_HEADS, SEQ / 128), 256, 0, stream>>>(
      qb, kb, vtb, ab);

  out_proj<<<blocks, 256, 0, stream>>>(ab, wob, (float*)d_out);
}
